// MultiHeadAttention_71803263254628
// MI455X (gfx1250) — matrix operand order repaired and checked by reading, not compiled, not run on hardware
//
#include <hip/hip_runtime.h>

typedef unsigned short ushort_t;
typedef unsigned int   uint_t;

typedef __attribute__((ext_vector_type(16))) __bf16 v16bf;
typedef __attribute__((ext_vector_type(8)))  float  v8f;

#define DM   1024   // d_model
#define NH   16     // heads
#define DK   64     // head dim
#define NB   4      // batch
#define SQ   2048   // seq len
#define NEG_INF (-3.0e38f)

union FragAB { v16bf v; uint4 q[2]; };

typedef __attribute__((address_space(3))) const void lds_cvoid;

__device__ __forceinline__ ushort_t f2bf(float f) {
  uint_t u = __float_as_uint(f);
  u += 0x7FFFu + ((u >> 16) & 1u);   // round to nearest even
  return (ushort_t)(u >> 16);
}

__device__ __forceinline__ v8f v8zero() {
  v8f z;
#pragma unroll
  for (int i = 0; i < 8; ++i) z[i] = 0.f;
  return z;
}

__device__ __forceinline__ v8f wmma_bf16(v16bf a, v16bf b, v8f c) {
  // D = A(16x32 bf16) * B(32x16 bf16) + C(16x16 f32)
  return __builtin_amdgcn_wmma_f32_16x16x32_bf16(
      /*neg_a=*/false, a, /*neg_b=*/false, b,
      /*c_mod=*/(short)0, c, /*reuse_a=*/false, /*reuse_b=*/false);
}

__device__ __forceinline__ void pack4(ushort_t* d, float4 f) {
  d[0] = f2bf(f.x); d[1] = f2bf(f.y); d[2] = f2bf(f.z); d[3] = f2bf(f.w);
}

// ---- CDNA5 async global->LDS copies (ASYNCcnt), ISA Ch.10 async ops ----
__device__ __forceinline__ unsigned lds_addr_of(const void* p) {
  return (unsigned)(size_t)(lds_cvoid*)p;   // addrspace(3) byte offset
}
__device__ __forceinline__ void async_copy_b128(void* lds_dst, const void* gsrc) {
  unsigned l = lds_addr_of(lds_dst);
  asm volatile("global_load_async_to_lds_b128 %0, %1, off"
               :: "v"(l), "v"(gsrc) : "memory");
}
__device__ __forceinline__ void async_copy_b128x2(void* lds_dst, const void* gsrc) {
  unsigned l = lds_addr_of(lds_dst);
  // INST_OFFSET applies to both LDS and global addresses (ISA 08_async_tensor §4.4)
  asm volatile("global_load_async_to_lds_b128 %0, %1, off\n\t"
               "global_load_async_to_lds_b128 %0, %1, off offset:16"
               :: "v"(l), "v"(gsrc) : "memory");
}
__device__ __forceinline__ void wait_async0() {
  asm volatile("s_wait_asynccnt 0" ::: "memory");
}

// ---------------------------------------------------------------------------
// fp32 -> bf16 converter (weights)
// ---------------------------------------------------------------------------
__global__ __launch_bounds__(256) void cvt_bf16_kernel(const float* __restrict__ x,
                                                       ushort_t* __restrict__ y, int n) {
  int i = (blockIdx.x * blockDim.x + threadIdx.x) * 4;
  if (i + 3 < n) {
    float4 f = *(const float4*)(x + i);
    union { uint2 d; ushort_t u[4]; } o;
    pack4(o.u, f);
    *(uint2*)(y + i) = o.d;
  }
}

// ---------------------------------------------------------------------------
// Tiled GEMM + bias:  Y[M,N] = X[M,K] @ W[N,K]^T + bias[N]
// W is pre-converted bf16 [N,K].  X: fp32 (IN_F32) or bf16.
// Y: fp32 (OUT_F32) or bf16.  Tile 128(M) x 64(N), K step 32, 8 waves.
// Double-buffered LDS; weight tiles staged with async global->LDS copies.
// ---------------------------------------------------------------------------
template <bool IN_F32, bool OUT_F32>
__global__ __launch_bounds__(256) void gemm_bias_kernel(
    const void* __restrict__ Xv, const ushort_t* __restrict__ Wb,
    const float* __restrict__ bias, void* __restrict__ Yv, int M, int N, int K) {
  __shared__ __align__(16) ushort_t At[2][128 * 32];
  __shared__ __align__(16) ushort_t Bt[2][64 * 32];

  const int tid = threadIdx.x;
  const int n0 = blockIdx.x * 64;
  const int m0 = blockIdx.y * 128;
  const int wid = tid >> 5, lane = tid & 31, lrow = lane & 15, half = lane >> 4;
  const int wm = (wid >> 1) * 32;   // wave M offset in tile (4 waves in M)
  const int wn = (wid & 1) * 32;    // wave N offset in tile (2 waves in N)

  const int arow = tid >> 1, ac16 = (tid & 1) * 16;  // A tile loader mapping
  const int brow = tid >> 2, bc8 = (tid & 3) * 8;    // B tile loader mapping

  v8f acc[2][2];
#pragma unroll
  for (int i = 0; i < 2; ++i)
#pragma unroll
    for (int j = 0; j < 2; ++j) acc[i][j] = v8zero();

  auto stage = [&](int buf, int kk) {
    // A tile [128 x 32]
    if (IN_F32) {
      const float* xp = (const float*)Xv + (size_t)(m0 + arow) * K + kk + ac16;
      float4 a0 = ((const float4*)xp)[0];
      float4 a1 = ((const float4*)xp)[1];
      float4 a2 = ((const float4*)xp)[2];
      float4 a3 = ((const float4*)xp)[3];
      union { uint4 q[2]; ushort_t u[16]; } o;
      pack4(o.u + 0, a0); pack4(o.u + 4, a1);
      pack4(o.u + 8, a2); pack4(o.u + 12, a3);
      *(uint4*)&At[buf][arow * 32 + ac16]     = o.q[0];
      *(uint4*)&At[buf][arow * 32 + ac16 + 8] = o.q[1];
    } else {
      const ushort_t* xp = (const ushort_t*)Xv + (size_t)(m0 + arow) * K + kk + ac16;
      async_copy_b128x2(&At[buf][arow * 32 + ac16], xp);
    }
    // B tile [64 x 32]: weight rows are K-contiguous -> pure async copy
    async_copy_b128(&Bt[buf][brow * 32 + bc8],
                    Wb + (size_t)(n0 + brow) * K + kk + bc8);
  };

  stage(0, 0);
  wait_async0();
  __syncthreads();

  int cur = 0;
  for (int kk = 0; kk < K; kk += 32) {
    if (kk + 32 < K) stage(cur ^ 1, kk + 32);   // overlap next tile with WMMAs

    FragAB af[2], bf[2];
#pragma unroll
    for (int i = 0; i < 2; ++i) {
      const ushort_t* base = &At[cur][(wm + i * 16 + lrow) * 32];
      af[i].q[0] = *(const uint4*)(base + half * 8);
      af[i].q[1] = *(const uint4*)(base + 16 + half * 8);
    }
#pragma unroll
    for (int j = 0; j < 2; ++j) {
      const ushort_t* base = &Bt[cur][(wn + j * 16 + lrow) * 32];
      bf[j].q[0] = *(const uint4*)(base + half * 8);
      bf[j].q[1] = *(const uint4*)(base + 16 + half * 8);
    }
#pragma unroll
    for (int i = 0; i < 2; ++i)
#pragma unroll
      for (int j = 0; j < 2; ++j) acc[i][j] = wmma_bf16(af[i].v, bf[j].v, acc[i][j]);

    wait_async0();
    __syncthreads();
    cur ^= 1;
  }

  // ---- epilogue: bias add + store ----
#pragma unroll
  for (int j = 0; j < 2; ++j) {
    const int col = n0 + wn + j * 16 + lrow;
    const float bv = bias[col];
#pragma unroll
    for (int i = 0; i < 2; ++i)
#pragma unroll
      for (int r = 0; r < 8; ++r) {
        const int row = m0 + wm + i * 16 + half * 8 + r;
        const float v = acc[i][j][r] + bv;
        if (OUT_F32) ((float*)Yv)[(size_t)row * N + col] = v;
        else         ((ushort_t*)Yv)[(size_t)row * N + col] = f2bf(v);
      }
  }
}

// ---------------------------------------------------------------------------
// Flash attention (causal).  One block = 128 query rows of one (b,h).
// 8 waves; each wave owns a 16-row stripe.  Streams 64-key blocks with
// double-buffered K/V tiles: async K copy + manual V transpose for block j+1
// overlap the 16 WMMAs + online softmax of block j (one barrier per block).
// ---------------------------------------------------------------------------
__global__ __launch_bounds__(256) void flash_attn_kernel(
    const ushort_t* __restrict__ Qb, const ushort_t* __restrict__ Kb,
    const ushort_t* __restrict__ Vb, ushort_t* __restrict__ Cb) {
  __shared__ __align__(16) ushort_t Kt[2][64 * 64];     // [key][dk]
  __shared__ __align__(16) ushort_t Vt[2][64 * 64];     // [dk][key] (transposed)
  __shared__ __align__(16) ushort_t Pst[8 * 16 * 64];   // wave-private P staging

  const int tid = threadIdx.x;
  const int q0 = blockIdx.x * 128;
  const int bh = blockIdx.y;
  const int b = bh >> 4, h = bh & 15;
  const int wid = tid >> 5, lane = tid & 31, lrow = lane & 15, half = lane >> 4;
  const int qrow0 = q0 + wid * 16;
  const size_t seqbase = (size_t)b * SQ;

  // Q fragments (16 rows x DK=64 -> two A frags), kept in registers
  FragAB qa[2];
  {
    const ushort_t* qp = Qb + (seqbase + qrow0 + lrow) * DM + h * DK;
#pragma unroll
    for (int f = 0; f < 2; ++f) {
      qa[f].q[0] = *(const uint4*)(qp + f * 32 + half * 8);
      qa[f].q[1] = *(const uint4*)(qp + f * 32 + 16 + half * 8);
    }
  }

  float m_r[8], l_r[8];
  v8f acc[4];
#pragma unroll
  for (int r = 0; r < 8; ++r) { m_r[r] = NEG_INF; l_r[r] = 0.f; }
#pragma unroll
  for (int s = 0; s < 4; ++s) acc[s] = v8zero();

  // cooperative loader mapping: 4 threads per key row, 16 dk each
  const int lkey = tid & 63, ldk = (tid >> 6) * 16;
  const int jend = q0 / 64 + 2;   // causal: keys up to q0+127

  auto stage = [&](int buf, int j) {
    const size_t krow = (seqbase + (size_t)j * 64 + lkey) * DM + h * DK + ldk;
    async_copy_b128x2(&Kt[buf][lkey * 64 + ldk], Kb + krow);
    const ushort_t* vg = Vb + krow;
    union { uint4 q[2]; ushort_t u[16]; } vv;
    vv.q[0] = ((const uint4*)vg)[0];
    vv.q[1] = ((const uint4*)vg)[1];
#pragma unroll
    for (int i = 0; i < 16; ++i) Vt[buf][(ldk + i) * 64 + lkey] = vv.u[i];
  };

  stage(0, 0);
  wait_async0();
  __syncthreads();

  int cur = 0;
  for (int j = 0; j < jend; ++j) {
    if (j + 1 < jend) stage(cur ^ 1, j + 1);   // overlap next block's staging

    // ---- scores S = Q K^T (16 x 64), scale + causal mask ----
    v8f s[4];
#pragma unroll
    for (int sub = 0; sub < 4; ++sub) {
      const ushort_t* kb = &Kt[cur][(sub * 16 + lrow) * 64];
      FragAB b0, b1;
      b0.q[0] = *(const uint4*)(kb + half * 8);
      b0.q[1] = *(const uint4*)(kb + 16 + half * 8);
      b1.q[0] = *(const uint4*)(kb + 32 + half * 8);
      b1.q[1] = *(const uint4*)(kb + 32 + 16 + half * 8);
      v8f c = v8zero();
      c = wmma_bf16(qa[0].v, b0.v, c);
      c = wmma_bf16(qa[1].v, b1.v, c);
      s[sub] = c;
    }
#pragma unroll
    for (int sub = 0; sub < 4; ++sub) {
      const int col = j * 64 + sub * 16 + lrow;
#pragma unroll
      for (int r = 0; r < 8; ++r) {
        const int row = qrow0 + half * 8 + r;
        const float v = s[sub][r] * 0.125f;  // 1/sqrt(64)
        s[sub][r] = (col > row) ? NEG_INF : v;
      }
    }

    // ---- online softmax: row max over this block (half-wave reduce) ----
    float bm[8];
#pragma unroll
    for (int r = 0; r < 8; ++r) {
      bm[r] = fmaxf(fmaxf(s[0][r], s[1][r]), fmaxf(s[2][r], s[3][r]));
#pragma unroll
      for (int off = 8; off >= 1; off >>= 1)
        bm[r] = fmaxf(bm[r], __shfl_xor(bm[r], off, 32));
    }

    ushort_t* pw = &Pst[wid * 16 * 64];
    float rs[8];
#pragma unroll
    for (int r = 0; r < 8; ++r) {
      const float mn = fmaxf(m_r[r], bm[r]);
      const float al = __expf(m_r[r] - mn);
      m_r[r] = mn;
      const int prow = half * 8 + r;
      float sum = 0.f;
#pragma unroll
      for (int sub = 0; sub < 4; ++sub) {
        const float p = __expf(s[sub][r] - mn);
        sum += p;
        pw[prow * 64 + sub * 16 + lrow] = f2bf(p);
        acc[sub][r] *= al;
      }
      l_r[r] = l_r[r] * al;
      rs[r] = sum;
    }
#pragma unroll
    for (int r = 0; r < 8; ++r) {
#pragma unroll
      for (int off = 8; off >= 1; off >>= 1) rs[r] += __shfl_xor(rs[r], off, 32);
      l_r[r] += rs[r];
    }

    // P staged via wave-private LDS (C-layout -> A-layout relayout)
    asm volatile("s_wait_dscnt 0" ::: "memory");
    FragAB pa[2];
    {
      const ushort_t* pb = pw + lrow * 64;
#pragma unroll
      for (int f = 0; f < 2; ++f) {
        pa[f].q[0] = *(const uint4*)(pb + f * 32 + half * 8);
        pa[f].q[1] = *(const uint4*)(pb + f * 32 + 16 + half * 8);
      }
    }

    // ---- O += P V  (keys = WMMA K-dim; V^T gives contiguous B frags) ----
#pragma unroll
    for (int sub = 0; sub < 4; ++sub) {
      const ushort_t* vb = &Vt[cur][(sub * 16 + lrow) * 64];
      FragAB b0, b1;
      b0.q[0] = *(const uint4*)(vb + half * 8);
      b0.q[1] = *(const uint4*)(vb + 16 + half * 8);
      b1.q[0] = *(const uint4*)(vb + 32 + half * 8);
      b1.q[1] = *(const uint4*)(vb + 32 + 16 + half * 8);
      acc[sub] = wmma_bf16(pa[0].v, b0.v, acc[sub]);
      acc[sub] = wmma_bf16(pa[1].v, b1.v, acc[sub]);
    }

    wait_async0();
    __syncthreads();
    cur ^= 1;
  }

  // ---- normalize + store context (bf16) ----
#pragma unroll
  for (int r = 0; r < 8; ++r) {
    const float inv = 1.0f / l_r[r];
    const int row = qrow0 + half * 8 + r;
    ushort_t* cg = Cb + (seqbase + row) * DM + h * DK;
#pragma unroll
    for (int sub = 0; sub < 4; ++sub)
      cg[sub * 16 + lrow] = f2bf(acc[sub][r] * inv);
  }
}

// ---------------------------------------------------------------------------
extern "C" void kernel_launch(void* const* d_in, const int* in_sizes, int n_in,
                              void* d_out, int out_size, void* d_ws, size_t ws_size,
                              hipStream_t stream) {
  (void)in_sizes; (void)n_in; (void)out_size; (void)ws_size;
  const float* query = (const float*)d_in[0];
  const float* key_  = (const float*)d_in[1];
  const float* value = (const float*)d_in[2];
  /* d_in[3] = mask: causal, computed analytically */
  const float* Wq = (const float*)d_in[4];  const float* bq = (const float*)d_in[5];
  const float* Wk = (const float*)d_in[6];  const float* bk = (const float*)d_in[7];
  const float* Wv = (const float*)d_in[8];  const float* bv = (const float*)d_in[9];
  const float* Wo = (const float*)d_in[10]; const float* bo = (const float*)d_in[11];
  float* out = (float*)d_out;

  // workspace layout (bf16 elements): 4 x 1M weights, 4 x 8M activations = 72 MB
  ushort_t* Wqb = (ushort_t*)d_ws;
  ushort_t* Wkb = Wqb + (1u << 20);
  ushort_t* Wvb = Wkb + (1u << 20);
  ushort_t* Wob = Wvb + (1u << 20);
  ushort_t* Qb  = Wob + (1u << 20);
  ushort_t* Kb  = Qb  + (8u << 20);
  ushort_t* Vb  = Kb  + (8u << 20);
  ushort_t* Cb  = Vb  + (8u << 20);

  const int M = NB * SQ;       // 8192
  const int wN = 1 << 20;      // elements per weight matrix

  dim3 cblk(256), cgrd(wN / (4 * 256));
  cvt_bf16_kernel<<<cgrd, cblk, 0, stream>>>(Wq, Wqb, wN);
  cvt_bf16_kernel<<<cgrd, cblk, 0, stream>>>(Wk, Wkb, wN);
  cvt_bf16_kernel<<<cgrd, cblk, 0, stream>>>(Wv, Wvb, wN);
  cvt_bf16_kernel<<<cgrd, cblk, 0, stream>>>(Wo, Wob, wN);

  dim3 ggrid(DM / 64, M / 128);  // (16, 64)
  gemm_bias_kernel<true, false><<<ggrid, 256, 0, stream>>>(query, Wqb, bq, Qb, M, DM, DM);
  gemm_bias_kernel<true, false><<<ggrid, 256, 0, stream>>>(key_,  Wkb, bk, Kb, M, DM, DM);
  gemm_bias_kernel<true, false><<<ggrid, 256, 0, stream>>>(value, Wvb, bv, Vb, M, DM, DM);

  dim3 agrid(SQ / 128, NB * NH);  // (16, 64)
  flash_attn_kernel<<<agrid, 256, 0, stream>>>(Qb, Kb, Vb, Cb);

  gemm_bias_kernel<false, true><<<ggrid, 256, 0, stream>>>(Cb, Wob, bo, out, M, DM, DM);
}
